// UnifiedEnergyFunction_76166950027540
// MI455X (gfx1250) — matrix operand order repaired and checked by reading, not compiled, not run on hardware
//
#include <hip/hip_runtime.h>
#include <math.h>

typedef __attribute__((ext_vector_type(16))) __bf16 v16bf;
typedef __attribute__((ext_vector_type(8)))  __bf16 v8bf;
typedef __attribute__((ext_vector_type(4)))  __bf16 v4bf;
typedef __attribute__((ext_vector_type(8)))  float  v8f;
typedef __attribute__((ext_vector_type(4)))  float  v4f;

#define B_ROWS 4096
#define DIM    256
#define NPAT   65536
#define BM     128                  // z-rows per block (8 waves x 16)
#define NSPLIT 64                   // N chunks
#define CHUNK  (NPAT / NSPLIT)      // 1024 patterns per block
#define TILES  (CHUNK / 16)         // 64 WMMA tiles per block
#define ZT_PAD 136                  // z-transpose row: 128 + 8 (272B, 16B aligned)
#define AM_PAD 264                  // M tile row: 256 + 8     (528B, 16B aligned)
#define LDS_ZT_ELEMS ((size_t)DIM * ZT_PAD)
#define LDS_AM_ELEMS ((size_t)2 * 16 * AM_PAD)
#define LDS_BYTES    ((LDS_ZT_ELEMS + LDS_AM_ELEMS) * 2)   // ~86.5 KB

// ---------------------------------------------------------------------------
// Fused GEMM (bf16 WMMA, f32 accum) + streaming logsumexp over the N axis.
//   A = M tile (16 patterns x K) streamed through a DOUBLE-BUFFERED LDS tile
//   (next tile's global loads issued before the WMMAs). All 8 A-fragments are
//   loaded into distinct registers BEFORE the WMMA chain; __launch_bounds__
//   (256,1) relaxes the VGPR cap so they actually stay live (pipelined dscnt
//   waits, no WAR v_nops). B = z tile, transposed once, pinned in VGPRs.
//   Logsumexp is lane-local (lane = z-row, VGPR index = pattern).
//   BF16M: stream a pre-converted bf16 copy of M (half the L2 bytes, no cvt).
// Grid: (B_ROWS/BM, NSPLIT), block: 256 threads (8 wave32).
// ---------------------------------------------------------------------------
template <bool BF16M>
__global__ __launch_bounds__(256, 1)
void energy_gemm_lse_kernel(const float* __restrict__ z,
                            const float* __restrict__ Mf32,
                            const __bf16* __restrict__ Mbf,
                            float* __restrict__ ws_lse)
{
    extern __shared__ __align__(16) __bf16 lds[];
    __bf16* Zt = lds;                     // [DIM][ZT_PAD]: z transposed [k][row]
    __bf16* Am = lds + LDS_ZT_ELEMS;      // [2][16][AM_PAD]: M tile, row-major

    const int tid    = threadIdx.x;       // 0..255
    const int lane   = tid & 31;
    const int wave   = tid >> 5;          // 0..7, each wave owns 16 z-rows
    const int row0   = blockIdx.x * BM;
    const int nchunk = blockIdx.y;

    // ---- transpose z tile once: 128 rows x 256 f32 -> Zt[k][row] bf16 ----
    {
        const int r0 = (tid & 31) * 4;
        const int d0 = (tid >> 5) * 32;
        const float* p0 = z + (size_t)(row0 + r0 + 0) * DIM + d0;
        const float* p1 = z + (size_t)(row0 + r0 + 1) * DIM + d0;
        const float* p2 = z + (size_t)(row0 + r0 + 2) * DIM + d0;
        const float* p3 = z + (size_t)(row0 + r0 + 3) * DIM + d0;
        #pragma unroll
        for (int j = 0; j < 8; ++j) {
            v4f f0 = *(const v4f*)(p0 + j * 4);
            v4f f1 = *(const v4f*)(p1 + j * 4);
            v4f f2 = *(const v4f*)(p2 + j * 4);
            v4f f3 = *(const v4f*)(p3 + j * 4);
            #pragma unroll
            for (int i = 0; i < 4; ++i) {
                v4bf p = { (__bf16)f0[i], (__bf16)f1[i], (__bf16)f2[i], (__bf16)f3[i] };
                *(v4bf*)(Zt + (size_t)(d0 + j * 4 + i) * ZT_PAD + r0) = p;
            }
        }
    }
    __syncthreads();

    // ---- preload the 8 z B-fragments (K = 8 x 32) per wave, pinned in VGPRs ----
    v16bf bfrag[8];
    #pragma unroll
    for (int kk = 0; kk < 8; ++kk) {
        const __bf16* bp = Zt + (size_t)(kk * 32 + ((lane >> 4) << 3)) * ZT_PAD + wave * 16 + (lane & 15);
        #pragma unroll
        for (int s = 0; s < 16; ++s)
            bfrag[kk][s] = bp[(size_t)((s & 7) + ((s >> 3) << 4)) * ZT_PAD];
    }

    float mrun = -INFINITY, srun = 0.0f;   // lane-local online logsumexp state

    const int nbase = nchunk * CHUNK;
    const int ahi   = (lane >> 4) << 3;    // +8 K offset for lanes 16..31
    const int amrow = lane & 15;           // A matrix row = pattern index
    const int fr    = tid >> 4;            // fill: pattern row 0..15
    const int fc0   = (tid & 15) * 16;     // fill: 16 elems per thread (coalesced)

    v4f  st[4];                            // f32 staging
    v8bf stb0, stb1;                       // bf16 staging

    // prologue: stage tile 0 into buffer 0
    if (BF16M) {
        const __bf16* src = Mbf + (size_t)(nbase + fr) * DIM + fc0;
        stb0 = *(const v8bf*)(src);
        stb1 = *(const v8bf*)(src + 8);
        __bf16* dst = Am + (size_t)fr * AM_PAD + fc0;
        *(v8bf*)(dst)     = stb0;
        *(v8bf*)(dst + 8) = stb1;
    } else {
        const float* src = Mf32 + (size_t)(nbase + fr) * DIM + fc0;
        #pragma unroll
        for (int j = 0; j < 4; ++j) st[j] = *(const v4f*)(src + j * 4);
        #pragma unroll
        for (int j = 0; j < 4; ++j)
            *(v4bf*)(Am + (size_t)fr * AM_PAD + fc0 + j * 4) =
                __builtin_convertvector(st[j], v4bf);
    }

    for (int t = 0; t < TILES; ++t) {
        __syncthreads();   // Am[t&1] visible; prior reads of Am[(t+1)&1] done

        // ---- issue next tile's global loads early (hidden behind WMMAs) ----
        if (t + 1 < TILES) {
            if (BF16M) {
                const __bf16* src = Mbf + (size_t)(nbase + (t + 1) * 16 + fr) * DIM + fc0;
                stb0 = *(const v8bf*)(src);
                stb1 = *(const v8bf*)(src + 8);
            } else {
                const float* src = Mf32 + (size_t)(nbase + (t + 1) * 16 + fr) * DIM + fc0;
                #pragma unroll
                for (int j = 0; j < 4; ++j) st[j] = *(const v4f*)(src + j * 4);
            }
        }

        // ---- load ALL 8 A-fragments into distinct registers (pipelined ds) ----
        const __bf16* amb = Am + (size_t)(t & 1) * 16 * AM_PAD
                               + (size_t)amrow * AM_PAD + ahi;
        v16bf afr[8];
        #pragma unroll
        for (int kk = 0; kk < 8; ++kk) {
            const __bf16* ap = amb + kk * 32;
            v8bf lo = *(const v8bf*)(ap);          // K = kk*32 + ahi + 0..7
            v8bf hi = *(const v8bf*)(ap + 16);     // K = kk*32 + ahi + 16..23
            afr[kk] = __builtin_shufflevector(lo, hi,
                0,1,2,3,4,5,6,7,8,9,10,11,12,13,14,15);
        }

        // ---- 8 chained WMMAs: tile [16 patterns x 16 z-rows], f32 accum ----
        v8f c = {};
        #pragma unroll
        for (int kk = 0; kk < 8; ++kk)
            c = __builtin_amdgcn_wmma_f32_16x16x32_bf16(
                    false, afr[kk], false, bfrag[kk], (short)0, c, false, false);

        // ---- store next tile into the other buffer ----
        if (t + 1 < TILES) {
            __bf16* dst = Am + (size_t)((t + 1) & 1) * 16 * AM_PAD
                             + (size_t)fr * AM_PAD + fc0;
            if (BF16M) {
                *(v8bf*)(dst)     = stb0;
                *(v8bf*)(dst + 8) = stb1;
            } else {
                #pragma unroll
                for (int j = 0; j < 4; ++j)
                    *(v4bf*)(dst + j * 4) = __builtin_convertvector(st[j], v4bf);
            }
        }

        // ---- lane-local online logsumexp over this tile's 8 patterns ----
        float t01 = fmaxf(c[0], c[1]), t23 = fmaxf(c[2], c[3]);
        float t45 = fmaxf(c[4], c[5]), t67 = fmaxf(c[6], c[7]);
        float tmax = fmaxf(fmaxf(t01, t23), fmaxf(t45, t67));
        float mnew = fmaxf(mrun, tmax);
        float s8 = __expf(c[0] - mnew) + __expf(c[1] - mnew)
                 + __expf(c[2] - mnew) + __expf(c[3] - mnew)
                 + __expf(c[4] - mnew) + __expf(c[5] - mnew)
                 + __expf(c[6] - mnew) + __expf(c[7] - mnew);
        srun = srun * __expf(mrun - mnew) + s8;
        mrun = mnew;
    }

    // ---- merge the two pattern halves (lane L <-> L+16), emit partial LSE ----
    {
        float mo = __shfl_xor(mrun, 16, 32);
        float so = __shfl_xor(srun, 16, 32);
        float mn = fmaxf(mrun, mo);
        float sn = srun * __expf(mrun - mn) + so * __expf(mo - mn);
        if (lane < 16) {
            const int row = row0 + wave * 16 + lane;
            ws_lse[(size_t)row * NSPLIT + nchunk] = mn + __logf(sn);
        }
    }
}

// ---------------------------------------------------------------------------
// One-time M f32 -> bf16 conversion into workspace (amortized over 32 reuses).
// ---------------------------------------------------------------------------
__global__ __launch_bounds__(256)
void convert_M_kernel(const float* __restrict__ M, __bf16* __restrict__ Mbf)
{
    const size_t i = ((size_t)blockIdx.x * 256 + threadIdx.x) * 8;
    v4f a = *(const v4f*)(M + i);
    v4f b = *(const v4f*)(M + i + 4);
    *(v4bf*)(Mbf + i)     = __builtin_convertvector(a, v4bf);
    *(v4bf*)(Mbf + i + 4) = __builtin_convertvector(b, v4bf);
}

// ---------------------------------------------------------------------------
// Combine NSPLIT partial LSEs per row -> sum of (-lse) into acc[0].
// ---------------------------------------------------------------------------
__global__ __launch_bounds__(256)
void row_lse_reduce_kernel(const float* __restrict__ ws_lse, float* __restrict__ acc)
{
    const int row = blockIdx.x * 256 + threadIdx.x;
    const float* p = ws_lse + (size_t)row * NSPLIT;
    float m = -INFINITY;
    #pragma unroll 8
    for (int i = 0; i < NSPLIT; ++i) m = fmaxf(m, p[i]);
    float s = 0.0f;
    #pragma unroll 8
    for (int i = 0; i < NSPLIT; ++i) s += __expf(p[i] - m);
    float contrib = -(m + __logf(s));

    __shared__ float red[256];
    red[threadIdx.x] = contrib;
    __syncthreads();
    for (int off = 128; off > 0; off >>= 1) {
        if (threadIdx.x < off) red[threadIdx.x] += red[threadIdx.x + off];
        __syncthreads();
    }
    if (threadIdx.x == 0) atomicAdd(&acc[0], red[0]);
}

// ---------------------------------------------------------------------------
// sum(z - z_next)^2 -> acc[1], sum(z^2) -> acc[2].
// ---------------------------------------------------------------------------
__global__ __launch_bounds__(256)
void vec_reduce_kernel(const float* __restrict__ z, const float* __restrict__ zn,
                       float* __restrict__ acc)
{
    const size_t total  = (size_t)B_ROWS * DIM;
    const size_t stride = (size_t)gridDim.x * 256;
    float d2 = 0.0f, z2 = 0.0f;
    for (size_t i = (size_t)blockIdx.x * 256 + threadIdx.x; i < total; i += stride) {
        float a = z[i], b = zn[i];
        float d = a - b;
        d2 += d * d;
        z2 += a * a;
    }
    __shared__ float r1[256], r2[256];
    r1[threadIdx.x] = d2; r2[threadIdx.x] = z2;
    __syncthreads();
    for (int off = 128; off > 0; off >>= 1) {
        if (threadIdx.x < off) {
            r1[threadIdx.x] += r1[threadIdx.x + off];
            r2[threadIdx.x] += r2[threadIdx.x + off];
        }
        __syncthreads();
    }
    if (threadIdx.x == 0) {
        atomicAdd(&acc[1], r1[0]);
        atomicAdd(&acc[2], r2[0]);
    }
}

__global__ void init_kernel(float* acc)
{
    if (threadIdx.x < 4) acc[threadIdx.x] = 0.0f;
}

__global__ void finalize_kernel(const float* __restrict__ acc, float* __restrict__ out)
{
    if (threadIdx.x == 0 && blockIdx.x == 0) {
        const float invB = 1.0f / (float)B_ROWS;
        float hop  = acc[0] * invB;                 // mean(-logsumexp)
        float cons = 1.0f  * acc[1] * invB;         // ALPHA = 1.0
        float reg  = 0.01f * acc[2] * invB;         // LAMBDA_L2 = 0.01
        out[0] = hop + cons + reg;
        out[1] = hop;
        out[2] = cons;
        out[3] = reg;
    }
}

extern "C" void kernel_launch(void* const* d_in, const int* in_sizes, int n_in,
                              void* d_out, int out_size, void* d_ws, size_t ws_size,
                              hipStream_t stream)
{
    const float* z  = (const float*)d_in[0];   // [4096, 256]
    const float* zn = (const float*)d_in[1];   // [4096, 256]
    const float* M  = (const float*)d_in[2];   // [65536, 256]
    float* out = (float*)d_out;                // 4 scalars
    float* acc    = (float*)d_ws;              // [0]=hop [1]=cons [2]=reg
    float* ws_lse = acc + 16;                  // [4096][64] partial LSEs (1 MB)

    // optional bf16 M cache after the LSE workspace (256B aligned)
    const size_t base    = (16 + (size_t)B_ROWS * NSPLIT) * sizeof(float);
    const size_t mbf_off = (base + 255) & ~(size_t)255;
    const size_t need_bf = mbf_off + (size_t)NPAT * DIM * sizeof(__bf16);
    __bf16* Mbf = (__bf16*)((char*)d_ws + mbf_off);
    const bool use_bf16m = (ws_size >= need_bf);   // fixed per harness -> deterministic

    init_kernel<<<1, 32, 0, stream>>>(acc);

    dim3 grid(B_ROWS / BM, NSPLIT);            // 32 x 64 = 2048 blocks
    if (use_bf16m) {
        convert_M_kernel<<<(NPAT * DIM) / (256 * 8), 256, 0, stream>>>(M, Mbf);
        energy_gemm_lse_kernel<true><<<grid, 256, LDS_BYTES, stream>>>(z, M, Mbf, ws_lse);
    } else {
        energy_gemm_lse_kernel<false><<<grid, 256, LDS_BYTES, stream>>>(z, M, Mbf, ws_lse);
    }

    row_lse_reduce_kernel<<<B_ROWS / 256, 256, 0, stream>>>(ws_lse, acc);
    vec_reduce_kernel<<<256, 256, 0, stream>>>(z, zn, acc);
    finalize_kernel<<<1, 32, 0, stream>>>(acc, out);
}
